// TransformerBlock_89215060672757
// MI455X (gfx1250) — matrix operand order repaired and checked by reading, not compiled, not run on hardware
//
#include <hip/hip_runtime.h>
#include <hip/hip_bf16.h>
#include <stdint.h>

#define HIDDEN   1024
#define HEADS    16
#define HEAD_DIM 64
#define INTER    4096
#define BATCH    2
#define SEQ      2048
#define BT       (BATCH*SEQ)

typedef __attribute__((ext_vector_type(16))) __bf16          v16bf;
typedef __attribute__((ext_vector_type(8)))  float           v8f;
typedef __attribute__((ext_vector_type(8)))  unsigned short  v8us;
typedef unsigned short u16;

union FragBF { v16bf v; v8us h[2]; u16 u[16]; };

__device__ __forceinline__ u16 f32_to_bf16(float f) {
    unsigned u = __float_as_uint(f);
    u += 0x7FFFu + ((u >> 16) & 1u);          // round-to-nearest-even
    return (u16)(u >> 16);
}
__device__ __forceinline__ float bf16_to_f32(u16 h) {
    return __uint_as_float(((unsigned)h) << 16);
}

// ---- CDNA5 async global->LDS (ASYNCcnt path, ISA §15.18 opc 98)
__device__ __forceinline__ void async_load_b128(const void* gaddr, void* laddr) {
    unsigned lds_off = (unsigned)(size_t)laddr;   // LDS generic addr: offset in [31:0]
    asm volatile("global_load_async_to_lds_b128 %0, %1, off"
                 :: "v"(lds_off), "v"(gaddr) : "memory");
}
__device__ __forceinline__ void wait_async() {
    asm volatile("s_wait_asynccnt 0x0" ::: "memory");
}

// ---- 16-lane DPP butterfly reductions (VALU, keeps LDS pipe free)
template <int CTRL>
__device__ __forceinline__ float dpp_movf(float v) {
    return __uint_as_float((unsigned)__builtin_amdgcn_update_dpp(
        0, (int)__float_as_uint(v), CTRL, 0xF, 0xF, true));
}
__device__ __forceinline__ float red_max16(float v) {
    v = fmaxf(v, dpp_movf<0xB1>(v));    // quad_perm(1,0,3,2) : xor 1
    v = fmaxf(v, dpp_movf<0x4E>(v));    // quad_perm(2,3,0,1) : xor 2
    v = fmaxf(v, dpp_movf<0x141>(v));   // row_half_mirror
    v = fmaxf(v, dpp_movf<0x140>(v));   // row_mirror
    return v;
}
__device__ __forceinline__ float red_sum16(float v) {
    v += dpp_movf<0xB1>(v);
    v += dpp_movf<0x4E>(v);
    v += dpp_movf<0x141>(v);
    v += dpp_movf<0x140>(v);
    return v;
}

// ---------------------------------------------------------------- elementwise
__global__ void k_f32_to_bf16(const float* __restrict__ in, u16* __restrict__ out, int n) {
    int i = blockIdx.x * blockDim.x + threadIdx.x;
    int stride = gridDim.x * blockDim.x;
    int n4 = n >> 2;
    const float4* in4 = (const float4*)in;
    ushort4* out4 = (ushort4*)out;
    for (; i < n4; i += stride) {
        float4 v = in4[i];
        ushort4 o;
        o.x = f32_to_bf16(v.x); o.y = f32_to_bf16(v.y);
        o.z = f32_to_bf16(v.z); o.w = f32_to_bf16(v.w);
        out4[i] = o;
    }
}

__global__ void k_silu_mul_bf16(const u16* __restrict__ g, const u16* __restrict__ u,
                                u16* __restrict__ h, int n) {
    int i = blockIdx.x * blockDim.x + threadIdx.x;
    int stride = gridDim.x * blockDim.x;
    int n4 = n >> 2;
    const ushort4* g4 = (const ushort4*)g;
    const ushort4* u4 = (const ushort4*)u;
    ushort4* h4 = (ushort4*)h;
    for (; i < n4; i += stride) {
        ushort4 gv = g4[i], uv = u4[i], o;
        float a;
        a = bf16_to_f32(gv.x); o.x = f32_to_bf16(a / (1.f + __expf(-a)) * bf16_to_f32(uv.x));
        a = bf16_to_f32(gv.y); o.y = f32_to_bf16(a / (1.f + __expf(-a)) * bf16_to_f32(uv.y));
        a = bf16_to_f32(gv.z); o.z = f32_to_bf16(a / (1.f + __expf(-a)) * bf16_to_f32(uv.z));
        a = bf16_to_f32(gv.w); o.w = f32_to_bf16(a / (1.f + __expf(-a)) * bf16_to_f32(uv.w));
        h4[i] = o;
    }
}

// ---------------------------------------------------------------- rmsnorm
__global__ __launch_bounds__(256) void k_rmsnorm_bf16(const float* __restrict__ x,
                                                      const float* __restrict__ w,
                                                      u16* __restrict__ out) {
    __shared__ float red[256];
    int row = blockIdx.x, tid = threadIdx.x;
    float4 v = ((const float4*)(x + (size_t)row * HIDDEN))[tid];
    red[tid] = v.x * v.x + v.y * v.y + v.z * v.z + v.w * v.w;
    __syncthreads();
    for (int o = 128; o > 0; o >>= 1) {
        if (tid < o) red[tid] += red[tid + o];
        __syncthreads();
    }
    float norm = rsqrtf(red[0] * (1.f / HIDDEN) + 1e-6f);
    float4 wv = ((const float4*)w)[tid];
    ushort4 o;
    o.x = f32_to_bf16(v.x * norm * wv.x);
    o.y = f32_to_bf16(v.y * norm * wv.y);
    o.z = f32_to_bf16(v.z * norm * wv.z);
    o.w = f32_to_bf16(v.w * norm * wv.w);
    ((ushort4*)(out + (size_t)row * HIDDEN))[tid] = o;
}

// ---------------------------------------------------------------- bf16 WMMA GEMM
template <int BF16_OUT>
__global__ __launch_bounds__(256) void k_gemm_bf16(
    const u16* __restrict__ A, const u16* __restrict__ B,
    float* __restrict__ Cf, u16* __restrict__ Cb, const float* __restrict__ resid,
    int M, int N, int K)
{
    __shared__ alignas(16) u16 sA [2][128][40];   // [m][k]  row stride 80B
    __shared__ alignas(16) u16 sBt[2][128][40];   // [n][k]  transposed

    const int tid  = threadIdx.x;
    const int wave = tid >> 5, lane = tid & 31;
    const int lrow = lane & 15, lhalf = lane >> 4;
    const int bm = blockIdx.y * 128, bn = blockIdx.x * 128;
    const int wm = (wave >> 1) * 32, wn = (wave & 1) * 64;

    const int ar0 = (tid * 2) >> 2,       ac0 = ((tid * 2) & 3) * 8;
    const int ar1 = (tid * 2 + 1) >> 2,   ac1 = ((tid * 2 + 1) & 3) * 8;

    auto stageA_async = [&](int k0, int buf) {
        async_load_b128(A + (size_t)(bm + ar0) * K + k0 + ac0, &sA[buf][ar0][ac0]);
        async_load_b128(A + (size_t)(bm + ar1) * K + k0 + ac1, &sA[buf][ar1][ac1]);
    };
    uint2 breg[4];
    auto loadB = [&](int k0) {
#pragma unroll
        for (int c = 0; c < 4; ++c) {
            int lin = (tid + c * 256) * 4;
            int r = lin >> 7, col = lin & 127;
            breg[c] = *(const uint2*)(B + (size_t)(k0 + r) * N + bn + col);
        }
    };
    auto storeB = [&](int buf) {
#pragma unroll
        for (int c = 0; c < 4; ++c) {
            int lin = (tid + c * 256) * 4;
            int r = lin >> 7, col = lin & 127;
            sBt[buf][col + 0][r] = (u16)(breg[c].x & 0xffff);
            sBt[buf][col + 1][r] = (u16)(breg[c].x >> 16);
            sBt[buf][col + 2][r] = (u16)(breg[c].y & 0xffff);
            sBt[buf][col + 3][r] = (u16)(breg[c].y >> 16);
        }
    };

    v8f zero = {};
    v8f acc[2][4];
#pragma unroll
    for (int a = 0; a < 2; ++a)
#pragma unroll
        for (int b2 = 0; b2 < 4; ++b2) acc[a][b2] = zero;

    stageA_async(0, 0);
    loadB(0);
    storeB(0);
    wait_async();
    __syncthreads();

    const int nk = K / 32;
    for (int kt = 0; kt < nk; ++kt) {
        const int buf = kt & 1;
        if (kt + 1 < nk) {
            stageA_async((kt + 1) * 32, buf ^ 1);   // async, overlaps compute
            loadB((kt + 1) * 32);                   // global loads in flight
        }

        FragBF af[2], bfg[4];
#pragma unroll
        for (int ms = 0; ms < 2; ++ms) {
            int row = wm + ms * 16 + lrow;
            af[ms].h[0] = *(const v8us*)&sA[buf][row][lhalf * 8];
            af[ms].h[1] = *(const v8us*)&sA[buf][row][16 + lhalf * 8];
        }
#pragma unroll
        for (int ns = 0; ns < 4; ++ns) {
            int col = wn + ns * 16 + lrow;
            bfg[ns].h[0] = *(const v8us*)&sBt[buf][col][lhalf * 8];
            bfg[ns].h[1] = *(const v8us*)&sBt[buf][col][16 + lhalf * 8];
        }
#pragma unroll
        for (int ms = 0; ms < 2; ++ms)
#pragma unroll
            for (int ns = 0; ns < 4; ++ns)
                acc[ms][ns] = __builtin_amdgcn_wmma_f32_16x16x32_bf16(
                    false, af[ms].v, false, bfg[ns].v, (short)0, acc[ms][ns], false, false);

        if (kt + 1 < nk) storeB(buf ^ 1);
        wait_async();
        __syncthreads();
    }

#pragma unroll
    for (int ms = 0; ms < 2; ++ms)
#pragma unroll
        for (int ns = 0; ns < 4; ++ns)
#pragma unroll
            for (int r = 0; r < 8; ++r) {
                int row = bm + wm + ms * 16 + lhalf * 8 + r;
                int col = bn + wn + ns * 16 + lrow;
                size_t idx = (size_t)row * N + col;
                float v = acc[ms][ns][r];
                if (BF16_OUT) {
                    Cb[idx] = f32_to_bf16(v);
                } else {
                    Cf[idx] = v + resid[idx];
                }
            }
}

// ---------------------------------------------------------------- flash attention
__global__ __launch_bounds__(256) void k_flash_attn(const u16* __restrict__ qkv,
                                                    u16* __restrict__ aout) {
    __shared__ alignas(16) u16 sK [32][72];    // [key][d]  row stride 144B
    __shared__ alignas(16) u16 sVt[64][40];    // [d][key]  transposed V block
    __shared__ alignas(16) u16 sP [8][16][40]; // per-wave P staging

    const int tid = threadIdx.x, wave = tid >> 5, lane = tid & 31;
    const int lrow = lane & 15, lhalf = lane >> 4;
    const int b = blockIdx.z, h = blockIdx.y;
    const int bx = gridDim.x - 1 - blockIdx.x;    // heaviest tiles launch first
    const int q0 = bx * 128 + wave * 16;
    const int kend = (bx + 1) * 128;              // uniform across the block
    const int RS = 3 * HIDDEN;
    const size_t base = (size_t)b * SEQ * RS + (size_t)h * HEAD_DIM;
    const u16* qp = qkv + base;
    const u16* kp = qkv + base + HIDDEN;
    const u16* vp = qkv + base + 2 * HIDDEN;

    FragBF qa[2];
    {
        const u16* qrp = qp + (size_t)(q0 + lrow) * RS;
#pragma unroll
        for (int f = 0; f < 2; ++f) {
            qa[f].h[0] = *(const v8us*)(qrp + f * 32 + lhalf * 8);
            qa[f].h[1] = *(const v8us*)(qrp + f * 32 + 16 + lhalf * 8);
        }
    }

    float m[8], l[8];
    v8f zero = {};
    v8f o[4];
#pragma unroll
    for (int r = 0; r < 8; ++r) { m[r] = -1e30f; l[r] = 0.f; }
#pragma unroll
    for (int t = 0; t < 4; ++t) o[t] = zero;
    const float scale = 0.125f;

    const int ckey = tid >> 3;             // cooperative copy: key row 0..31
    const int cd   = (tid & 7) * 8;        // d chunk

    for (int kb = 0; kb < kend; kb += 32) {
        // K block async (layout-preserving); V block manual transpose
        async_load_b128(kp + (size_t)(kb + ckey) * RS + cd, &sK[ckey][cd]);
        {
            const u16* src = vp + (size_t)(kb + ckey) * RS + cd;
#pragma unroll
            for (int j = 0; j < 8; ++j) sVt[cd + j][ckey] = src[j];
        }
        if (kb + 32 < kend)
            __builtin_prefetch(vp + (size_t)(kb + 32 + ckey) * RS + cd, 0, 3);
        wait_async();
        __syncthreads();

        // wave-uniform: skip compute for blocks entirely above the diagonal
        if (kb <= q0 + 15) {
            // S = Q @ K^T
            FragBF kf[2][2];
#pragma unroll
            for (int nt = 0; nt < 2; ++nt) {
                int key = nt * 16 + lrow;
#pragma unroll
                for (int s = 0; s < 2; ++s) {
                    kf[nt][s].h[0] = *(const v8us*)&sK[key][s * 32 + lhalf * 8];
                    kf[nt][s].h[1] = *(const v8us*)&sK[key][s * 32 + 16 + lhalf * 8];
                }
            }
            v8f sf[2];
#pragma unroll
            for (int nt = 0; nt < 2; ++nt) {
                sf[nt] = zero;
#pragma unroll
                for (int s = 0; s < 2; ++s)
                    sf[nt] = __builtin_amdgcn_wmma_f32_16x16x32_bf16(
                        false, qa[s].v, false, kf[nt][s].v, (short)0, sf[nt], false, false);
            }

            // online softmax; mask only on the diagonal block (wave-uniform branch)
            const bool needmask = (kb + 31 > q0);
            float pv[2][8];
#pragma unroll
            for (int r = 0; r < 8; ++r) {
                float v0 = sf[0][r] * scale;
                float v1 = sf[1][r] * scale;
                if (needmask) {
                    int q = q0 + lhalf * 8 + r;
                    if (kb + lrow      > q) v0 = -1e30f;
                    if (kb + 16 + lrow > q) v1 = -1e30f;
                }
                float bmax = red_max16(fmaxf(v0, v1));
                float mn = fmaxf(m[r], bmax);
                float corr = __expf(m[r] - mn);
                m[r] = mn;
                float p0 = __expf(v0 - mn);
                float p1 = __expf(v1 - mn);
                pv[0][r] = p0;
                pv[1][r] = p1;
                float psum = red_sum16(p0 + p1);
                l[r] = l[r] * corr + psum;
#pragma unroll
                for (int t = 0; t < 4; ++t) o[t][r] *= corr;
            }

            // P (C-layout) -> per-wave LDS -> A-frag
#pragma unroll
            for (int nt = 0; nt < 2; ++nt)
#pragma unroll
                for (int r = 0; r < 8; ++r)
                    sP[wave][lhalf * 8 + r][nt * 16 + lrow] = f32_to_bf16(pv[nt][r]);
            FragBF pa;
            pa.h[0] = *(const v8us*)&sP[wave][lrow][lhalf * 8];
            pa.h[1] = *(const v8us*)&sP[wave][lrow][16 + lhalf * 8];

            // O += P @ V
#pragma unroll
            for (int dt = 0; dt < 4; ++dt) {
                int d = dt * 16 + lrow;
                FragBF vf;
                vf.h[0] = *(const v8us*)&sVt[d][lhalf * 8];
                vf.h[1] = *(const v8us*)&sVt[d][16 + lhalf * 8];
                o[dt] = __builtin_amdgcn_wmma_f32_16x16x32_bf16(
                    false, pa.v, false, vf.v, (short)0, o[dt], false, false);
            }
        }
        __syncthreads();   // protect sK/sVt before next block's staging
    }

#pragma unroll
    for (int dt = 0; dt < 4; ++dt)
#pragma unroll
        for (int r = 0; r < 8; ++r) {
            int q = q0 + lhalf * 8 + r;
            float v = o[dt][r] / l[r];
            aout[((size_t)b * SEQ + q) * HIDDEN + h * HEAD_DIM + dt * 16 + lrow] =
                f32_to_bf16(v);
        }
}

// ---------------------------------------------------------------- host
extern "C" void kernel_launch(void* const* d_in, const int* in_sizes, int n_in,
                              void* d_out, int out_size, void* d_ws, size_t ws_size,
                              hipStream_t stream) {
    const float* x      = (const float*)d_in[0];
    const float* anw    = (const float*)d_in[1];
    const float* fnw    = (const float*)d_in[2];
    const float* w_qkv  = (const float*)d_in[3];
    const float* w_out  = (const float*)d_in[4];
    const float* w_gate = (const float*)d_in[5];
    const float* w_up   = (const float*)d_in[6];
    const float* w_down = (const float*)d_in[7];
    float* out = (float*)d_out;

    char* p = (char*)d_ws;
    auto alloc = [&](size_t bytes) { char* r = p; p += (bytes + 255) & ~(size_t)255; return r; };

    u16* wqkv_b  = (u16*)alloc((size_t)HIDDEN * 3 * HIDDEN * 2);
    u16* wout_b  = (u16*)alloc((size_t)HIDDEN * HIDDEN * 2);
    u16* wgate_b = (u16*)alloc((size_t)HIDDEN * INTER * 2);
    u16* wup_b   = (u16*)alloc((size_t)HIDDEN * INTER * 2);
    u16* wdown_b = (u16*)alloc((size_t)INTER * HIDDEN * 2);
    u16* xn_b    = (u16*)alloc((size_t)BT * HIDDEN * 2);
    u16* qkv_b   = (u16*)alloc((size_t)BT * 3 * HIDDEN * 2);
    u16* attn_b  = (u16*)alloc((size_t)BT * HIDDEN * 2);
    float* x1    = (float*)alloc((size_t)BT * HIDDEN * 4);
    u16* g_b     = (u16*)alloc((size_t)BT * INTER * 2);
    u16* u_b     = (u16*)alloc((size_t)BT * INTER * 2);
    u16* h_b     = (u16*)alloc((size_t)BT * INTER * 2);

    k_f32_to_bf16<<<1024, 256, 0, stream>>>(w_qkv,  wqkv_b,  HIDDEN * 3 * HIDDEN);
    k_f32_to_bf16<<<1024, 256, 0, stream>>>(w_out,  wout_b,  HIDDEN * HIDDEN);
    k_f32_to_bf16<<<1024, 256, 0, stream>>>(w_gate, wgate_b, HIDDEN * INTER);
    k_f32_to_bf16<<<1024, 256, 0, stream>>>(w_up,   wup_b,   HIDDEN * INTER);
    k_f32_to_bf16<<<1024, 256, 0, stream>>>(w_down, wdown_b, INTER * HIDDEN);

    k_rmsnorm_bf16<<<BT, 256, 0, stream>>>(x, anw, xn_b);

    k_gemm_bf16<1><<<dim3(3 * HIDDEN / 128, BT / 128), 256, 0, stream>>>(
        xn_b, wqkv_b, nullptr, qkv_b, nullptr, BT, 3 * HIDDEN, HIDDEN);

    k_flash_attn<<<dim3(SEQ / 128, HEADS, BATCH), 256, 0, stream>>>(qkv_b, attn_b);

    k_gemm_bf16<0><<<dim3(HIDDEN / 128, BT / 128), 256, 0, stream>>>(
        attn_b, wout_b, x1, nullptr, x, BT, HIDDEN, HIDDEN);

    k_rmsnorm_bf16<<<BT, 256, 0, stream>>>(x1, fnw, xn_b);

    k_gemm_bf16<1><<<dim3(INTER / 128, BT / 128), 256, 0, stream>>>(
        xn_b, wgate_b, nullptr, g_b, nullptr, BT, INTER, HIDDEN);
    k_gemm_bf16<1><<<dim3(INTER / 128, BT / 128), 256, 0, stream>>>(
        xn_b, wup_b, nullptr, u_b, nullptr, BT, INTER, HIDDEN);

    k_silu_mul_bf16<<<1024, 256, 0, stream>>>(g_b, u_b, h_b, BT * INTER);

    k_gemm_bf16<0><<<dim3(HIDDEN / 128, BT / 128), 256, 0, stream>>>(
        h_b, wdown_b, out, nullptr, x1, BT, HIDDEN, INTER);
}
